// Attention_7395933684041
// MI455X (gfx1250) — compile-verified, matrix operand order checked by reading, not run on hardware
//
#include <hip/hip_runtime.h>
#include <hip/hip_bf16.h>

typedef __bf16 v8bf  __attribute__((ext_vector_type(8)));
typedef __bf16 v16bf __attribute__((ext_vector_type(16)));
typedef float  v8f   __attribute__((ext_vector_type(8)));

#define Bn 2
#define Nn 2048
#define Cn 1024
#define Hn 16
#define Dn 64
#define CAPF 30.0f

__device__ __forceinline__ v8f zero8() {
    v8f z = {0.f,0.f,0.f,0.f,0.f,0.f,0.f,0.f};
    return z;
}

// 16x32(K) bf16 fragment: lane holds two contiguous 8-elem chunks at
// K = (lane>>4)*8 and +16 (ISA 7.12.2 16-bit layout) -> two b128 loads.
__device__ __forceinline__ v16bf load_frag(const __bf16* p) {
    v8bf lo = *(const v8bf*)(p);
    v8bf hi = *(const v8bf*)(p + 16);
    return __builtin_shufflevector(lo, hi, 0,1,2,3,4,5,6,7,8,9,10,11,12,13,14,15);
}

__device__ __forceinline__ v8f wmma_bf16(v16bf a, v16bf b, v8f c) {
    return __builtin_amdgcn_wmma_f32_16x16x32_bf16(false, a, false, b, (short)0, c, false, false);
}

// CDNA5 V_TANH_F32 (transcendental VALU) -- single trans op.
__device__ __forceinline__ float fast_tanh(float x) {
    float r;
    asm volatile("v_tanh_f32 %0, %1" : "=v"(r) : "v"(x));
    return r;
}

// p = exp(CAP * tanh(s / CAP)) : one v_tanh_f32 + one v_exp_f32
__device__ __forceinline__ float softcap_exp(float s) {
    return __expf(CAPF * fast_tanh(s * (1.0f / CAPF)));
}

// ---------------------------------------------------------------- converts
__global__ void cvt_f32_bf16(const float* __restrict__ src,
                             __bf16* __restrict__ dst, int n) {
    int i = blockIdx.x * blockDim.x + threadIdx.x;
    if (i < n) dst[i] = (__bf16)src[i];
}

// ---------------------------------------------------------- QKV GEMM + LN
// Wave tile 32(M) x 64(N). Ping-pong double buffer; loop pinned at its
// natural 2-phase unroll so the allocator doesn't split accumulators.
__global__ __launch_bounds__(256)
void qkv_gemm_ln(const __bf16* __restrict__ xb, const __bf16* __restrict__ wb,
                 const float* __restrict__ qkv_b,
                 const float* __restrict__ qn_w, const float* __restrict__ qn_b,
                 const float* __restrict__ kn_w, const float* __restrict__ kn_b,
                 __bf16* __restrict__ qb, __bf16* __restrict__ kb,
                 __bf16* __restrict__ vt) {
    const int wave = threadIdx.x >> 5;
    const int lane = threadIdx.x & 31;
    const int lh   = lane & 15;
    const int hf   = lane >> 4;
    const int mbase = blockIdx.y * 256 + wave * 32;
    const int obase = blockIdx.x * 64;

    const __bf16* ar[2];
    const __bf16* br[4];
    ar[0] = xb + (long)(mbase +      lh) * Cn;
    ar[1] = xb + (long)(mbase + 16 + lh) * Cn;
    #pragma unroll
    for (int t = 0; t < 4; t++) br[t] = wb + (long)(obase + t * 16 + lh) * Cn;

    v8f acc[2][4];
    #pragma unroll
    for (int i = 0; i < 2; i++)
        #pragma unroll
        for (int t = 0; t < 4; t++) acc[i][t] = zero8();

    v16bf af[2][2], bf[2][4];
    auto LD = [&](int p, int kk) {
        int ko = kk + hf * 8;
        af[p][0] = load_frag(ar[0] + ko);
        af[p][1] = load_frag(ar[1] + ko);
        #pragma unroll
        for (int t = 0; t < 4; t++) bf[p][t] = load_frag(br[t] + ko);
    };
    auto MM = [&](int p) {
        #pragma unroll
        for (int i = 0; i < 2; i++)
            #pragma unroll
            for (int t = 0; t < 4; t++)
                acc[i][t] = wmma_bf16(af[p][i], bf[p][t], acc[i][t]);
    };

    LD(0, 0);
    #pragma unroll 1
    for (int kk = 0; kk + 64 < Cn; kk += 64) {
        LD(1, kk + 32);
        MM(0);
        LD(0, kk + 64);
        MM(1);
    }
    LD(1, Cn - 32);
    MM(0);
    MM(1);

    // bias (per output column)
    float bia[4];
    #pragma unroll
    for (int t = 0; t < 4; t++) bia[t] = qkv_b[obase + t * 16 + lh];
    #pragma unroll
    for (int i = 0; i < 2; i++)
        #pragma unroll
        for (int t = 0; t < 4; t++) acc[i][t] = acc[i][t] + bia[t];

    const int section = obase / Cn;          // 0=q, 1=k, 2=v
    const int h = (obase % Cn) / Dn;

    if (section < 2) {
        const float* lw = (section == 0) ? qn_w : kn_w;
        const float* lb = (section == 0) ? qn_b : kn_b;
        float w[4], bw[4];
        #pragma unroll
        for (int t = 0; t < 4; t++) { w[t] = lw[t*16+lh]; bw[t] = lb[t*16+lh]; }
        const float qscale = (section == 0) ? 0.125f : 1.0f;   // D^-0.5
        #pragma unroll
        for (int i = 0; i < 2; i++) {
            #pragma unroll
            for (int j = 0; j < 8; j++) {
                float x0 = acc[i][0][j], x1 = acc[i][1][j];
                float x2 = acc[i][2][j], x3 = acc[i][3][j];
                float s  = x0 + x1 + x2 + x3;
                float s2 = x0*x0 + x1*x1 + x2*x2 + x3*x3;
                #pragma unroll
                for (int m = 1; m < 16; m <<= 1) {
                    s  += __shfl_xor(s,  m, 16);
                    s2 += __shfl_xor(s2, m, 16);
                }
                float mean = s * (1.0f / 64.0f);
                float var  = s2 * (1.0f / 64.0f) - mean * mean;
                float rstd = rsqrtf(var + 1e-5f);
                acc[i][0][j] = ((x0 - mean) * rstd * w[0] + bw[0]) * qscale;
                acc[i][1][j] = ((x1 - mean) * rstd * w[1] + bw[1]) * qscale;
                acc[i][2][j] = ((x2 - mean) * rstd * w[2] + bw[2]) * qscale;
                acc[i][3][j] = ((x3 - mean) * rstd * w[3] + bw[3]) * qscale;
            }
        }
    }

    // scatter (C layout: VGPR j -> row j (lanes 0-15), j+8 (lanes 16-31))
    #pragma unroll
    for (int i = 0; i < 2; i++) {
        #pragma unroll
        for (int j = 0; j < 8; j++) {
            int r  = mbase + i * 16 + j + 8 * hf;
            int bb = r >> 11;
            int n  = r & 2047;
            long base = ((long)(bb * Hn + h) * Nn + n) * Dn;
            if (section == 0) {
                #pragma unroll
                for (int t = 0; t < 4; t++)
                    qb[base + t*16 + lh] = (__bf16)acc[i][t][j];
            } else if (section == 1) {
                #pragma unroll
                for (int t = 0; t < 4; t++)
                    kb[base + t*16 + lh] = (__bf16)acc[i][t][j];
            } else {
                long vb = (long)(bb * Hn + h) * Dn * Nn + n;
                #pragma unroll
                for (int t = 0; t < 4; t++)
                    vt[vb + (long)(t*16 + lh) * Nn] = (__bf16)acc[i][t][j];
            }
        }
    }
}

// -------------------------------------------------------------- attention
// Per wave 16 query rows. Keys streamed in 32-blocks, ping-pong K-fragment
// prefetch; V loads issued early so tanh/exp VALU overlaps vmem.
__global__ __launch_bounds__(256)
void attn_kernel(const __bf16* __restrict__ qb, const __bf16* __restrict__ kb,
                 const __bf16* __restrict__ vt, __bf16* __restrict__ ob) {
    __shared__ __bf16 Pld[8 * 16 * 32];

    const int wave = threadIdx.x >> 5;
    const int lane = threadIdx.x & 31;
    const int lh   = lane & 15;
    const int hf   = lane >> 4;
    const int bh = blockIdx.y;
    const int b  = bh / Hn, h = bh % Hn;
    const int qbase = blockIdx.x * 128 + wave * 16;

    const long head = (long)(b * Hn + h);
    const __bf16* qrow = qb + (head * Nn + qbase + lh) * Dn;
    v16bf aq0 = load_frag(qrow +      hf * 8);   // d 0..31
    v16bf aq1 = load_frag(qrow + 32 + hf * 8);   // d 32..63

    v8f o[4];
    #pragma unroll
    for (int t = 0; t < 4; t++) o[t] = zero8();
    v8f dn = zero8();

    __bf16* pbase = Pld + wave * 16 * 32;
    const __bf16* kbh = kb + head * Nn * Dn;
    const __bf16* vth = vt + head * Dn * Nn;

    v16bf kf[2][4];
    auto LDK = [&](int p, int jj) {
        const __bf16* kr0 = kbh + (long)(jj +      lh) * Dn;
        const __bf16* kr1 = kbh + (long)(jj + 16 + lh) * Dn;
        kf[p][0] = load_frag(kr0 +      hf * 8);
        kf[p][1] = load_frag(kr0 + 32 + hf * 8);
        kf[p][2] = load_frag(kr1 +      hf * 8);
        kf[p][3] = load_frag(kr1 + 32 + hf * 8);
    };

    auto STEP = [&](int p, int jcur, int jnext) {
        v16bf vf[4];
        #pragma unroll
        for (int t = 0; t < 4; t++)
            vf[t] = load_frag(vth + (long)(t*16 + lh) * Nn + jcur + hf * 8);

        v8f s0 = zero8(), s1 = zero8();
        s0 = wmma_bf16(aq0, kf[p][0], s0);
        s0 = wmma_bf16(aq1, kf[p][1], s0);
        s1 = wmma_bf16(aq0, kf[p][2], s1);
        s1 = wmma_bf16(aq1, kf[p][3], s1);

        LDK(p ^ 1, jnext);   // prefetch next K block into other buffer

        #pragma unroll
        for (int jv = 0; jv < 8; jv++) {
            float p0 = softcap_exp(s0[jv]);
            float p1 = softcap_exp(s1[jv]);
            dn[jv] += p0 + p1;
            int r = jv + 8 * hf;
            pbase[r * 32 + lh]      = (__bf16)p0;
            pbase[r * 32 + lh + 16] = (__bf16)p1;
        }
        asm volatile("s_wait_dscnt 0" ::: "memory");

        v16bf pf = load_frag(pbase + lh * 32 + hf * 8);
        #pragma unroll
        for (int t = 0; t < 4; t++) o[t] = wmma_bf16(pf, vf[t], o[t]);
    };

    LDK(0, 0);
    #pragma unroll 1
    for (int j = 0; j < Nn; j += 64) {
        STEP(0, j,      j + 32);
        STEP(1, j + 32, (j + 64) & (Nn - 1));
    }

    // normalize (row denominator across 16 lanes of one half-wave)
    #pragma unroll
    for (int jv = 0; jv < 8; jv++) {
        float s = dn[jv];
        #pragma unroll
        for (int m = 1; m < 16; m <<= 1) s += __shfl_xor(s, m, 16);
        float inv = 1.0f / s;
        #pragma unroll
        for (int t = 0; t < 4; t++) o[t][jv] *= inv;
    }

    // store to (B, N, C) bf16
    #pragma unroll
    for (int jv = 0; jv < 8; jv++) {
        int n = qbase + jv + 8 * hf;
        long base = ((long)(b * Nn + n)) * Cn + h * Dn;
        #pragma unroll
        for (int t = 0; t < 4; t++)
            ob[base + t*16 + lh] = (__bf16)o[t][jv];
    }
}

// -------------------------------------------------------------- proj GEMM
// Wave tile 32 x 64, ping-pong pipelined; fp32 output + bias.
__global__ __launch_bounds__(256)
void proj_gemm(const __bf16* __restrict__ ab, const __bf16* __restrict__ wb,
               const float* __restrict__ bias, float* __restrict__ out) {
    const int wave = threadIdx.x >> 5;
    const int lane = threadIdx.x & 31;
    const int lh   = lane & 15;
    const int hf   = lane >> 4;
    const int mbase = blockIdx.y * 256 + wave * 32;
    const int obase = blockIdx.x * 64;

    const __bf16* ar[2];
    const __bf16* br[4];
    ar[0] = ab + (long)(mbase +      lh) * Cn;
    ar[1] = ab + (long)(mbase + 16 + lh) * Cn;
    #pragma unroll
    for (int t = 0; t < 4; t++) br[t] = wb + (long)(obase + t * 16 + lh) * Cn;

    v8f acc[2][4];
    #pragma unroll
    for (int i = 0; i < 2; i++)
        #pragma unroll
        for (int t = 0; t < 4; t++) acc[i][t] = zero8();

    v16bf af[2][2], bf[2][4];
    auto LD = [&](int p, int kk) {
        int ko = kk + hf * 8;
        af[p][0] = load_frag(ar[0] + ko);
        af[p][1] = load_frag(ar[1] + ko);
        #pragma unroll
        for (int t = 0; t < 4; t++) bf[p][t] = load_frag(br[t] + ko);
    };
    auto MM = [&](int p) {
        #pragma unroll
        for (int i = 0; i < 2; i++)
            #pragma unroll
            for (int t = 0; t < 4; t++)
                acc[i][t] = wmma_bf16(af[p][i], bf[p][t], acc[i][t]);
    };

    LD(0, 0);
    #pragma unroll 1
    for (int kk = 0; kk + 64 < Cn; kk += 64) {
        LD(1, kk + 32);
        MM(0);
        LD(0, kk + 64);
        MM(1);
    }
    LD(1, Cn - 32);
    MM(0);
    MM(1);

    float bi[4];
    #pragma unroll
    for (int t = 0; t < 4; t++) bi[t] = bias[obase + t * 16 + lh];

    #pragma unroll
    for (int i = 0; i < 2; i++)
        #pragma unroll
        for (int j = 0; j < 8; j++) {
            long r = mbase + i * 16 + j + 8 * hf;
            #pragma unroll
            for (int t = 0; t < 4; t++)
                out[r * Cn + obase + t*16 + lh] = acc[i][t][j] + bi[t];
        }
}

// ------------------------------------------------------------------ launch
extern "C" void kernel_launch(void* const* d_in, const int* in_sizes, int n_in,
                              void* d_out, int out_size, void* d_ws, size_t ws_size,
                              hipStream_t stream) {
    const float* x      = (const float*)d_in[0];
    const float* qkv_w  = (const float*)d_in[1];
    const float* qkv_b  = (const float*)d_in[2];
    const float* qn_w   = (const float*)d_in[3];
    const float* qn_b   = (const float*)d_in[4];
    const float* kn_w   = (const float*)d_in[5];
    const float* kn_b   = (const float*)d_in[6];
    const float* proj_w = (const float*)d_in[7];
    const float* proj_b = (const float*)d_in[8];
    float* out = (float*)d_out;

    const long nX  = (long)Bn * Nn * Cn;        // 4194304
    const long nW  = 3L * Cn * Cn;              // 3145728
    const long nPW = (long)Cn * Cn;             // 1048576
    const long nHD = (long)Bn * Hn * Nn * Dn;   // 4194304

    char* ws = (char*)d_ws;
    __bf16* xb  = (__bf16*)ws;                 ws += nX  * 2;
    __bf16* wb  = (__bf16*)ws;                 ws += nW  * 2;
    __bf16* pwb = (__bf16*)ws;                 ws += nPW * 2;
    __bf16* qb  = (__bf16*)ws;                 ws += nHD * 2;
    __bf16* kb  = (__bf16*)ws;                 ws += nHD * 2;
    __bf16* vt  = (__bf16*)ws;                 ws += nHD * 2;
    __bf16* ob  = (__bf16*)ws;                 ws += nX  * 2;

    cvt_f32_bf16<<<(nX  + 255) / 256, 256, 0, stream>>>(x,      xb,  (int)nX);
    cvt_f32_bf16<<<(nW  + 255) / 256, 256, 0, stream>>>(qkv_w,  wb,  (int)nW);
    cvt_f32_bf16<<<(nPW + 255) / 256, 256, 0, stream>>>(proj_w, pwb, (int)nPW);

    qkv_gemm_ln<<<dim3(3 * Cn / 64, Bn * Nn / 256), 256, 0, stream>>>(
        xb, wb, qkv_b, qn_w, qn_b, kn_w, kn_b, qb, kb, vt);

    attn_kernel<<<dim3(Nn / 128, Bn * Hn), 256, 0, stream>>>(qb, kb, vt, ob);

    proj_gemm<<<dim3(Cn / 64, Bn * Nn / 256), 256, 0, stream>>>(ob, pwb, proj_b, out);
}
